// TanhFixedPointLayer_1219770712654
// MI455X (gfx1250) — matrix operand order repaired and checked by reading, not compiled, not run on hardware
//
#include <hip/hip_runtime.h>

#define D_DIM 512
#define B_DIM 65536
#define TOTAL (B_DIM * D_DIM)   // 33,554,432 elements
#define KTILE 32
#define ROWS_WG 128
#define LDS_STRIDE 40           // 32 k-elements + 8 pad (80B row stride, conflict-free b128)
#define KSTEPS (D_DIM / KTILE)  // 16

typedef __attribute__((ext_vector_type(16))) __bf16 v16bf;
typedef __attribute__((ext_vector_type(8)))  __bf16 v8bf;
typedef __attribute__((ext_vector_type(8)))  float  v8f;

typedef int v4i __attribute__((vector_size(16)));
typedef __attribute__((address_space(1))) v4i as1_v4i;
typedef __attribute__((address_space(3))) v4i as3_v4i;

#if __has_builtin(__builtin_amdgcn_global_load_async_to_lds_b128)
#define HAVE_ASYNC_B128 1
#else
#define HAVE_ASYNC_B128 0
#endif

__device__ __forceinline__ void wait_async_done() {
#if HAVE_ASYNC_B128
#if __has_builtin(__builtin_amdgcn_s_wait_asynccnt)
    __builtin_amdgcn_s_wait_asynccnt(0);
#else
    asm volatile("s_wait_asynccnt 0" ::: "memory");
#endif
#endif
}

__device__ __forceinline__ void async_copy_b128(const __bf16* gsrc, __bf16* ldst) {
#if HAVE_ASYNC_B128
    __builtin_amdgcn_global_load_async_to_lds_b128(
        (as1_v4i*)(void*)gsrc, (as3_v4i*)(void*)ldst, 0, 0);
#else
    *(v8bf*)ldst = *(const v8bf*)gsrc;   // global_load_b128 + ds_store_b128 fallback
#endif
}

__device__ __forceinline__ float act_tanh(float v) {
#if __has_builtin(__builtin_amdgcn_tanhf)
    return __builtin_amdgcn_tanhf(v);
#elif __has_builtin(__builtin_amdgcn_tanh_f32)
    return __builtin_amdgcn_tanh_f32(v);
#else
    return tanhf(v);
#endif
}

// ---------------------------------------------------------------------------
// W (fp32 [D,D]) -> split bf16 planes Wh (RNE high) / Wl (residual)
// ---------------------------------------------------------------------------
__global__ void prep_w_kernel(const float* __restrict__ W,
                              __bf16* __restrict__ Wh,
                              __bf16* __restrict__ Wl) {
    int i = blockIdx.x * blockDim.x + threadIdx.x;
    float w = W[i];
    __bf16 h = (__bf16)w;
    Wh[i] = h;
    Wl[i] = (__bf16)(w - (float)h);
}

// ---------------------------------------------------------------------------
// Iteration 1: z1 = tanh(x), written as split-bf16 planes
// ---------------------------------------------------------------------------
__global__ void tanh0_kernel(const float4* __restrict__ x,
                             __bf16* __restrict__ zh, __bf16* __restrict__ zl) {
    int i = blockIdx.x * blockDim.x + threadIdx.x;
    float4 v = x[i];
    float f[4] = {v.x, v.y, v.z, v.w};
#pragma unroll
    for (int t = 0; t < 4; ++t) {
        float z = act_tanh(f[t]);
        __bf16 h = (__bf16)z;
        zh[i * 4 + t] = h;
        zl[i * 4 + t] = (__bf16)(z - (float)h);
    }
}

// ---------------------------------------------------------------------------
// One step: out = tanh(z @ W^T + x), split-bf16 WMMA: zh*Wh + zh*Wl + zl*Wh.
// A tiles staged in LDS (async, double-buffered); B frags from L2-resident W.
// WG: 256 threads = 8 waves (2 wave-rows x 4 wave-cols); wave tile 64x32;
// WG tile 128x128; grid (D/128, B/128) = (4, 512).
// ---------------------------------------------------------------------------
__global__ void __launch_bounds__(256)
fixed_point_step_kernel(const __bf16* __restrict__ zh_in,
                        const __bf16* __restrict__ zl_in,
                        const float*  __restrict__ x,
                        const __bf16* __restrict__ Wh,
                        const __bf16* __restrict__ Wl,
                        __bf16* __restrict__ zh_out,
                        __bf16* __restrict__ zl_out,
                        float*  __restrict__ f32_out,
                        int write_f32) {
    __shared__ __attribute__((aligned(16))) __bf16 smem[2][2][ROWS_WG * LDS_STRIDE];

    const int tid    = threadIdx.x;
    const int lane   = tid & 31;
    const int lane16 = lane & 15;
    const int sel    = lane >> 4;         // 0: lanes 0-15, 1: lanes 16-31
    const int wave   = tid >> 5;
    const int wrow   = wave >> 2;         // 0..1 (64 rows each)
    const int wcol   = wave & 3;          // 0..3 (32 cols each)

    const int blk_row  = blockIdx.y * 128;
    const int base_col = blockIdx.x * 128 + wcol * 32;

    // Cooperative stage loader: 1024 x b128 chunks (2 planes x 128 rows x 4),
    // 4 chunks per thread.
    auto stage_load = [&](int stage, int k0) {
#pragma unroll
        for (int i = 0; i < 4; ++i) {
            int c     = tid + i * 256;
            int plane = c >> 9;           // 0 = hi, 1 = lo
            int rc    = c & 511;
            int row   = rc >> 2;          // 0..127
            int kc    = rc & 3;           // 8-element (16B) chunk within k-tile
            const __bf16* g = (plane ? zl_in : zh_in)
                              + (size_t)(blk_row + row) * D_DIM + k0 + kc * 8;
            __bf16* l = &smem[stage][plane][row * LDS_STRIDE + kc * 8];
            async_copy_b128(g, l);
        }
    };

    v8f acc[4][2];
#pragma unroll
    for (int m = 0; m < 4; ++m)
#pragma unroll
        for (int n = 0; n < 2; ++n)
#pragma unroll
            for (int j = 0; j < 8; ++j)
                acc[m][n][j] = 0.0f;

    stage_load(0, 0);

    for (int ks = 0; ks < KSTEPS; ++ks) {
        wait_async_done();                // own async stores to LDS complete
        __syncthreads();                  // => everyone's stage `ks` data visible
        if (ks + 1 < KSTEPS)
            stage_load((ks + 1) & 1, (ks + 1) * KTILE);

        const __bf16* sh = smem[ks & 1][0];
        const __bf16* sl = smem[ks & 1][1];

        // A frags (16x32 bf16): lane holds row lane%16; V0-3 = K base..base+7,
        // V4-7 = K base+8..base+15, base = sel*16  -> two 16B ds loads each.
        v16bf ah[4], al[4];
#pragma unroll
        for (int m = 0; m < 4; ++m) {
            int rl = wrow * 64 + m * 16 + lane16;
            const __bf16* ph = sh + rl * LDS_STRIDE;
            const __bf16* pl = sl + rl * LDS_STRIDE;
            v8bf h0 = *(const v8bf*)(ph + sel * 16);
            v8bf h1 = *(const v8bf*)(ph + 8 + sel * 16);
            v8bf l0 = *(const v8bf*)(pl + sel * 16);
            v8bf l1 = *(const v8bf*)(pl + 8 + sel * 16);
            ah[m] = __builtin_shufflevector(h0, h1, 0, 1, 2, 3, 4, 5, 6, 7,
                                            8, 9, 10, 11, 12, 13, 14, 15);
            al[m] = __builtin_shufflevector(l0, l1, 0, 1, 2, 3, 4, 5, 6, 7,
                                            8, 9, 10, 11, 12, 13, 14, 15);
        }

        // B frags (32x16 bf16): lane holds col lane%16, K = sel*16..sel*16+15
        // contiguous from W row (L2 resident).
        const int k0 = ks * KTILE;
#pragma unroll
        for (int n = 0; n < 2; ++n) {
            const size_t woff =
                (size_t)(base_col + n * 16 + lane16) * D_DIM + k0 + sel * 16;
            v16bf bh = *(const v16bf*)(Wh + woff);
            v16bf bl = *(const v16bf*)(Wl + woff);
#pragma unroll
            for (int m = 0; m < 4; ++m) {
                acc[m][n] = __builtin_amdgcn_wmma_f32_16x16x32_bf16(
                    false, ah[m], false, bh, (short)0, acc[m][n], false, false);
                acc[m][n] = __builtin_amdgcn_wmma_f32_16x16x32_bf16(
                    false, ah[m], false, bl, (short)0, acc[m][n], false, false);
                acc[m][n] = __builtin_amdgcn_wmma_f32_16x16x32_bf16(
                    false, al[m], false, bh, (short)0, acc[m][n], false, false);
            }
        }
    }

    // Fused epilogue: z = tanh(acc + x); write split-bf16 planes, or fp32 on
    // the final iteration. C layout: VGPR j -> row m*16 + sel*8 + j, col n*16+lane16.
#pragma unroll
    for (int m = 0; m < 4; ++m) {
#pragma unroll
        for (int n = 0; n < 2; ++n) {
#pragma unroll
            for (int j = 0; j < 8; ++j) {
                int row = blk_row + wrow * 64 + m * 16 + sel * 8 + j;
                int col = base_col + n * 16 + lane16;
                size_t idx = (size_t)row * D_DIM + col;
                float z = act_tanh(acc[m][n][j] + x[idx]);
                if (write_f32) {
                    f32_out[idx] = z;
                } else {
                    __bf16 h = (__bf16)z;
                    zh_out[idx] = h;
                    zl_out[idx] = (__bf16)(z - (float)h);
                }
            }
        }
    }
}

extern "C" void kernel_launch(void* const* d_in, const int* in_sizes, int n_in,
                              void* d_out, int out_size, void* d_ws, size_t ws_size,
                              hipStream_t stream) {
    const float* x = (const float*)d_in[0];  // [B, D] fp32
    const float* W = (const float*)d_in[1];  // [D, D] fp32

    // Workspace: plane set A (zh, zl) | Wh | Wl        (128 MB + 1 MB)
    __bf16* zhA = (__bf16*)d_ws;
    __bf16* zlA = zhA + (size_t)TOTAL;
    __bf16* Wh  = zlA + (size_t)TOTAL;
    __bf16* Wl  = Wh + (size_t)D_DIM * D_DIM;
    // Plane set B reuses d_out's 128 MB (overwritten with fp32 at iteration 50).
    __bf16* zhB = (__bf16*)d_out;
    __bf16* zlB = zhB + (size_t)TOTAL;

    prep_w_kernel<<<(D_DIM * D_DIM) / 256, 256, 0, stream>>>(W, Wh, Wl);
    tanh0_kernel<<<TOTAL / (256 * 4), 256, 0, stream>>>((const float4*)x, zhA, zlA);

    __bf16 *cur_h = zhA, *cur_l = zlA, *nxt_h = zhB, *nxt_l = zlB;
    dim3 grid(D_DIM / 128, B_DIM / 128);
    for (int it = 2; it <= 50; ++it) {
        int wf = (it == 50) ? 1 : 0;     // iteration 50 reads set A, writes fp32
        fixed_point_step_kernel<<<grid, 256, 0, stream>>>(
            cur_h, cur_l, x, Wh, Wl, nxt_h, nxt_l, (float*)d_out, wf);
        __bf16* t;
        t = cur_h; cur_h = nxt_h; nxt_h = t;
        t = cur_l; cur_l = nxt_l; nxt_l = t;
    }
}
